// SelfAttention_21981642621024
// MI455X (gfx1250) — matrix operand order repaired and checked by reading, not compiled, not run on hardware
//
#include <hip/hip_runtime.h>

// ---------------- problem constants ----------------
#define Bb   2
#define Ss   2048
#define Dd   2048
#define Hh   16
#define DHh  128
#define EPSf 1e-6f

typedef __attribute__((ext_vector_type(16))) _Float16 v16h;
typedef __attribute__((ext_vector_type(8)))  _Float16 v8h;
typedef __attribute__((ext_vector_type(8)))  float    v8f;

// A-fragment (16x32 f16, M=lane&15): halves[0:8] at K=khalf.., halves[8:16] at K=16+khalf..
__device__ __forceinline__ v16h frag_two(const _Float16* p0, const _Float16* p1) {
  union { v16h h; v8h q[2]; } u;
  u.q[0] = *reinterpret_cast<const v8h*>(p0);
  u.q[1] = *reinterpret_cast<const v8h*>(p1);
  return u.h;
}
// contiguous-K loader (16 halves from p, k = 16h + s); unused, B-fragments use frag_two
__device__ __forceinline__ v16h frag_cont(const _Float16* p) {
  union { v16h h; v8h q[2]; } u;
  u.q[0] = *reinterpret_cast<const v8h*>(p);
  u.q[1] = *reinterpret_cast<const v8h*>(p + 8);
  return u.h;
}
__device__ __forceinline__ v8f wmma16(v16h a, v16h b, v8f c) {
  return __builtin_amdgcn_wmma_f32_16x16x32_f16(false, a, false, b, (short)0, c, false, false);
}

// ---------------- f32 -> f16 convert ----------------
__global__ void cvt_f16(const float* __restrict__ in, _Float16* __restrict__ out, int n) {
  int i = blockIdx.x * blockDim.x + threadIdx.x;
  if (i < n) out[i] = (_Float16)in[i];
}

// ---------------- GEMM: Out[M,N] = A[M,K] * Bw[N,K]^T + bias ----------------
// wave tile: 32(M) x 128(N); block = 4 waves covering 32 x 512.
template <bool HALF_OUT>
__global__ __launch_bounds__(128) void gemm_f16(const _Float16* __restrict__ A,
                                                const _Float16* __restrict__ Bw,
                                                const float* __restrict__ bias,
                                                void* __restrict__ Out,
                                                int M, int N, int K) {
  const int lane  = threadIdx.x & 31;
  const int wave  = threadIdx.x >> 5;
  const int lm    = lane & 15;
  const int khalf = (lane >> 4) * 8;    // A-frag K offset (0/8), also C-layout M offset
  const int bhalf = (lane >> 4) * 16;   // unused; B-frags use khalf too
  const int mbase = blockIdx.x * 32;
  const int nbase = (blockIdx.y * 4 + wave) * 128;

  v8f acc0[8], acc1[8];
#pragma unroll
  for (int t = 0; t < 8; ++t) { acc0[t] = (v8f){}; acc1[t] = (v8f){}; }

  const _Float16* Arow0 = A + (size_t)(mbase + lm) * K;
  const _Float16* Arow1 = A + (size_t)(mbase + 16 + lm) * K;
  for (int k0 = 0; k0 < K; k0 += 32) {
    if (k0 + 256 < K) {  // gfx1250 global_prefetch_b8 one tile-row ahead
      __builtin_prefetch(Arow0 + k0 + 256, 0, 1);
      __builtin_prefetch(Arow1 + k0 + 256, 0, 1);
    }
    v16h af0 = frag_two(Arow0 + k0 + khalf, Arow0 + k0 + 16 + khalf);
    v16h af1 = frag_two(Arow1 + k0 + khalf, Arow1 + k0 + 16 + khalf);
#pragma unroll
    for (int t = 0; t < 8; ++t) {
      const _Float16* Brow = Bw + (size_t)(nbase + t * 16 + lm) * K;
      v16h bf = frag_two(Brow + k0 + khalf, Brow + k0 + 16 + khalf);
      acc0[t] = wmma16(af0, bf, acc0[t]);
      acc1[t] = wmma16(af1, bf, acc1[t]);
    }
  }
#pragma unroll
  for (int t = 0; t < 8; ++t) {
    int n = nbase + t * 16 + lm;
    float bv = bias[n];
#pragma unroll
    for (int r = 0; r < 8; ++r) {
      int m0 = mbase + r + khalf;
      float v0 = acc0[t][r] + bv;
      float v1 = acc1[t][r] + bv;
      if (HALF_OUT) {
        ((_Float16*)Out)[(size_t)m0 * N + n]        = (_Float16)v0;
        ((_Float16*)Out)[(size_t)(m0 + 16) * N + n] = (_Float16)v1;
      } else {
        ((float*)Out)[(size_t)m0 * N + n]        = v0;
        ((float*)Out)[(size_t)(m0 + 16) * N + n] = v1;
      }
    }
  }
}

// ---------------- RoPE + per-head RMS + layout change ----------------
// qkv_h: [B,S,3*D] f16.  Writes q_h (pre-scaled by 1/sqrt(DH)), k_h: [B,H,S,DH]; vt: [B,H,DH,S].
__global__ __launch_bounds__(128) void rope_rms_prep(const _Float16* __restrict__ qkv,
                                                     const float* __restrict__ rp,    // [S, DH/2]
                                                     const float* __restrict__ qk_w,  // [DH]
                                                     _Float16* __restrict__ qh,
                                                     _Float16* __restrict__ kh,
                                                     _Float16* __restrict__ vt) {
  const int lane = threadIdx.x & 31;
  const int wave = threadIdx.x >> 5;
  const int s = blockIdx.x * 4 + wave;
  const int h = blockIdx.y, b = blockIdx.z;
  const int dh0 = lane * 4;

  const size_t base = ((size_t)(b * Ss + s)) * (3 * Dd) + h * DHh + dh0;
  float q[4], k[4], v[4];
#pragma unroll
  for (int j = 0; j < 4; ++j) {
    q[j] = (float)qkv[base + j];
    k[j] = (float)qkv[base + Dd + j];
    v[j] = (float)qkv[base + 2 * Dd + j];
  }
  // RoPE on pairs (0,1),(2,3)
#pragma unroll
  for (int p = 0; p < 2; ++p) {
    float ang = rp[(size_t)s * (DHh / 2) + lane * 2 + p];
    float cs = __cosf(ang), sn = __sinf(ang);
    float q0 = q[2 * p], q1 = q[2 * p + 1];
    q[2 * p] = q0 * cs - q1 * sn;  q[2 * p + 1] = q0 * sn + q1 * cs;
    float k0 = k[2 * p], k1 = k[2 * p + 1];
    k[2 * p] = k0 * cs - k1 * sn;  k[2 * p + 1] = k0 * sn + k1 * cs;
  }
  // RMS over DH=128 (32 lanes x 4)
  float sq = 0.f, sk = 0.f;
#pragma unroll
  for (int j = 0; j < 4; ++j) { sq += q[j] * q[j]; sk += k[j] * k[j]; }
#pragma unroll
  for (int off = 16; off > 0; off >>= 1) {
    sq += __shfl_xor(sq, off, 32);
    sk += __shfl_xor(sk, off, 32);
  }
  // fold softmax scale 1/sqrt(DH) into q
  float rq = rsqrtf(sq / DHh + EPSf) * 0.08838834764831845f;
  float rk = rsqrtf(sk / DHh + EPSf);

  const size_t qb = ((size_t)((b * Hh + h) * Ss + s)) * DHh + dh0;
  const size_t vb = ((size_t)(b * Hh + h)) * DHh * Ss;
#pragma unroll
  for (int j = 0; j < 4; ++j) {
    float w = qk_w[dh0 + j];
    qh[qb + j] = (_Float16)(q[j] * rq * w);
    kh[qb + j] = (_Float16)(k[j] * rk * w);
    vt[vb + (size_t)(dh0 + j) * Ss + s] = (_Float16)v[j];
  }
}

// ---------------- flash attention step (S^T = K x Q^T, O^T += V^T x P^T) ----------------
template <bool MASKED>
__device__ __forceinline__ void attn_step(int kv0, int kvEnd, int qrow,
                                          int lm, int khalf, bool lowhalf,
                                          const _Float16* __restrict__ Khead,
                                          const _Float16* __restrict__ Vhead,
                                          const v16h* qf, v8f* o,
                                          float& m_i, float& l_i) {
  const bool t1valid = !MASKED || (kv0 + 16) < kvEnd;
  v8f st0 = (v8f){}, st1 = (v8f){};
  const _Float16* Kr0 = Khead + (size_t)(kv0 + lm) * DHh;
  const _Float16* Kr1 = Khead + (size_t)((t1valid ? kv0 + 16 : kv0) + lm) * DHh;
#pragma unroll
  for (int c = 0; c < 4; ++c) {
    v16h a0 = frag_two(Kr0 + c * 32 + khalf, Kr0 + c * 32 + 16 + khalf);
    st0 = wmma16(a0, qf[c], st0);
    v16h a1 = frag_two(Kr1 + c * 32 + khalf, Kr1 + c * 32 + 16 + khalf);
    st1 = wmma16(a1, qf[c], st1);
  }
  // online softmax; lane owns query row qrow, holds kv rows {r+khalf} per tile
  float s0[8], s1[8];
  float mx = m_i;
#pragma unroll
  for (int r = 0; r < 8; ++r) {
    float va = st0[r];
    float vb = st1[r];
    if (MASKED) {
      if (kv0 + r + khalf > qrow) va = -1e30f;
      if (!t1valid || (kv0 + 16 + r + khalf > qrow)) vb = -1e30f;
    }
    s0[r] = va; s1[r] = vb;
    mx = fmaxf(mx, fmaxf(va, vb));
  }
  mx = fmaxf(mx, __shfl_xor(mx, 16, 32));
  float alpha = __expf(m_i - mx);
  float rsum = 0.f;
  float p0[8], p1[8];
#pragma unroll
  for (int r = 0; r < 8; ++r) {
    p0[r] = __expf(s0[r] - mx);
    p1[r] = __expf(s1[r] - mx);
    rsum += p0[r] + p1[r];
  }
  rsum += __shfl_xor(rsum, 16, 32);
  l_i = l_i * alpha + rsum;
  m_i = mx;
  // P^T B-fragment: lanes<16 need kv 0..7,16..23, lanes>=16 need kv 8..15,24..31
  v16h pb;
#pragma unroll
  for (int r = 0; r < 8; ++r) {
    float q0 = __shfl_xor(p0[r], 16, 32);  // partner's tile0 (rows ^8)
    float q1 = __shfl_xor(p1[r], 16, 32);  // partner's tile1
    pb[r]     = (_Float16)p0[r];
    pb[8 + r] = (_Float16)p1[r];
  }
  const int k1off = t1valid ? kv0 + 16 : kv0;  // clamp OOB tile-1 loads (P==0 there)
#pragma unroll
  for (int c = 0; c < 8; ++c) {
#pragma unroll
    for (int r = 0; r < 8; ++r) o[c][r] *= alpha;
    const _Float16* Vrow = Vhead + (size_t)(c * 16 + lm) * Ss;
    v16h vf = frag_two(Vrow + kv0 + khalf, Vrow + k1off + khalf);
    o[c] = wmma16(vf, pb, o[c]);
  }
}

// ---------------- causal flash attention, 1 wave per 16 query rows ----------------
__global__ __launch_bounds__(32) void flash_attn(const _Float16* __restrict__ qh,
                                                 const _Float16* __restrict__ kh,
                                                 const _Float16* __restrict__ vt,
                                                 float* __restrict__ out) {
  const int lane  = threadIdx.x & 31;
  const int lm    = lane & 15;
  const int khalf = (lane >> 4) * 8;
  const int bhalf = (lane >> 4) * 16;
  const bool lowhalf = lane < 16;
  const int qbase = blockIdx.x * 16;
  const int h = blockIdx.y, b = blockIdx.z;
  const int qrow = qbase + lm;

  const size_t head = ((size_t)(b * Hh + h)) * Ss * DHh;
  const _Float16* Qrow = qh + head + (size_t)qrow * DHh;
  v16h qf[4];  // B-fragments of Q^T over the 4 dh-chunks of 32
#pragma unroll
  for (int c = 0; c < 4; ++c) qf[c] = frag_two(Qrow + c * 32 + khalf, Qrow + c * 32 + 16 + khalf);

  v8f o[8];  // O^T: 8 tiles of (16 dh x 16 q), f32
#pragma unroll
  for (int t = 0; t < 8; ++t) o[t] = (v8f){};
  float m_i = -1e30f, l_i = 0.f;

  const int kvEnd   = qbase + 16;
  const int fullEnd = qbase & ~31;  // kv0 < fullEnd guarantees every kv <= qrow for all lanes
  const _Float16* Khead = kh + head;
  const _Float16* Vhead = vt + head;  // [DH, S]

  int kv0 = 0;
  for (; kv0 < fullEnd; kv0 += 32)
    attn_step<false>(kv0, kvEnd, qrow, lm, khalf, lowhalf, Khead, Vhead, qf, o, m_i, l_i);
  for (; kv0 < kvEnd; kv0 += 32)
    attn_step<true>(kv0, kvEnd, qrow, lm, khalf, lowhalf, Khead, Vhead, qf, o, m_i, l_i);

  // finalize: lane owns query row; write pre-norm f32 [B,S,D]
  float inv_l = 1.f / l_i;
  const size_t ob = ((size_t)(b * Ss + qrow)) * Dd + h * DHh;
#pragma unroll
  for (int c = 0; c < 8; ++c)
#pragma unroll
    for (int r = 0; r < 8; ++r)
      out[ob + c * 16 + r + khalf] = o[c][r] * inv_l;
}

// ---------------- output RMS norm: a_h = rmsnorm(a)*w, f16 ----------------
__global__ __launch_bounds__(256) void rms_out(const float* __restrict__ a,
                                               const float* __restrict__ w,
                                               _Float16* __restrict__ ah) {
  __shared__ float red[8];
  const int tid = threadIdx.x;
  const size_t row = blockIdx.x;
  const float* ap = a + row * Dd;
  float vals[8];
  float s = 0.f;
#pragma unroll
  for (int j = 0; j < 8; ++j) {
    vals[j] = ap[tid + j * 256];
    s += vals[j] * vals[j];
  }
#pragma unroll
  for (int off = 16; off > 0; off >>= 1) s += __shfl_xor(s, off, 32);
  if ((tid & 31) == 0) red[tid >> 5] = s;
  __syncthreads();
  float tot = 0.f;
#pragma unroll
  for (int i = 0; i < 8; ++i) tot += red[i];
  float r = rsqrtf(tot / Dd + EPSf);
#pragma unroll
  for (int j = 0; j < 8; ++j)
    ah[row * Dd + tid + j * 256] = (_Float16)(vals[j] * r * w[tid + j * 256]);
}

// ---------------- driver ----------------
extern "C" void kernel_launch(void* const* d_in, const int* in_sizes, int n_in,
                              void* d_out, int out_size, void* d_ws, size_t ws_size,
                              hipStream_t stream) {
  (void)in_sizes; (void)n_in; (void)out_size; (void)ws_size;
  const float* x     = (const float*)d_in[0];
  const float* rp    = (const float*)d_in[1];
  const float* Win   = (const float*)d_in[2];
  const float* b_in  = (const float*)d_in[3];
  const float* Wout  = (const float*)d_in[4];
  const float* b_out = (const float*)d_in[5];
  const float* qk_w  = (const float*)d_in[6];
  const float* out_w = (const float*)d_in[7];
  float* out = (float*)d_out;

  char* ws = (char*)d_ws;
  size_t off = 0;
  auto halloc = [&](size_t elems) {
    _Float16* p = (_Float16*)(ws + off);
    off += ((elems * sizeof(_Float16) + 255) / 256) * 256;
    return p;
  };
  const size_t BSD = (size_t)Bb * Ss * Dd;  // 8,388,608
  _Float16* x_h    = halloc(BSD);
  _Float16* win_h  = halloc((size_t)3 * Dd * Dd);
  _Float16* wout_h = halloc((size_t)Dd * Dd);
  _Float16* qkv_h  = halloc((size_t)Bb * Ss * 3 * Dd);
  _Float16* q_h    = halloc(BSD);
  _Float16* k_h    = halloc(BSD);
  _Float16* vt_h   = halloc(BSD);
  _Float16* a_h    = halloc(BSD);

  const int M = Bb * Ss;  // 4096
  cvt_f16<<<(int)((BSD + 255) / 256), 256, 0, stream>>>(x, x_h, (int)BSD);
  cvt_f16<<<(3 * Dd * Dd + 255) / 256, 256, 0, stream>>>(Win, win_h, 3 * Dd * Dd);
  cvt_f16<<<(Dd * Dd + 255) / 256, 256, 0, stream>>>(Wout, wout_h, Dd * Dd);

  // qkv = x @ Win^T + b_in  (f16 out)
  dim3 g1(M / 32, (3 * Dd) / 512);
  gemm_f16<true><<<g1, 128, 0, stream>>>(x_h, win_h, b_in, qkv_h, M, 3 * Dd, Dd);

  dim3 gp(Ss / 4, Hh, Bb);
  rope_rms_prep<<<gp, 128, 0, stream>>>(qkv_h, rp, qk_w, q_h, k_h, vt_h);

  dim3 ga(Ss / 16, Hh, Bb);
  flash_attn<<<ga, 32, 0, stream>>>(q_h, k_h, vt_h, out);  // pre-norm attn staged in d_out

  rms_out<<<Bb * Ss, 256, 0, stream>>>(out, out_w, a_h);

  // final = a_h @ Wout^T + b_out (f32 out)
  dim3 g2(M / 32, Dd / 512);
  gemm_f16<false><<<g2, 128, 0, stream>>>(a_h, wout_h, b_out, out, M, Dd, Dd);
}
